// BCAblock_Anchor_5823975653914
// MI455X (gfx1250) — matrix operand order repaired and checked by reading, not compiled, not run on hardware
//
#include <hip/hip_runtime.h>
#include <hip/hip_bf16.h>
#include <math.h>

// ---------------------------------------------------------------------------
// BCAblock_Anchor for MI455X (gfx1250, wave32, WMMA).
//   GEMMs (q / kv0 / kv1 / proj / fc1+GELU / fc2) -> v_wmma_f32_16x16x32_bf16.
//   Register tiling: each wave computes a 16x64 output tile (4 accumulators),
//   A fragment loaded once per K-step (2 x b128) and reused across 4 B tiles
//   (packed WMMA-fragment order, 2 x b128 each).  K loop fully unrolled.
//   Window attention (81 disp, HC=32 dots, softmax, bilateral gather) -> VALU,
//   online softmax, fully register resident.
// ---------------------------------------------------------------------------

typedef __attribute__((ext_vector_type(16))) __bf16 v16bf;
typedef __attribute__((ext_vector_type(8)))  float  v8f;

namespace {
constexpr int kB    = 2;
constexpr int kC    = 128;
constexpr int kNH   = 4;
constexpr int kHC   = 32;
constexpr int kH    = 192;
constexpr int kW    = 192;
constexpr int kNpix = kH * kW;          // 36864
constexpr int kBN   = kB * kNpix;       // 73728 rows in the token stream
constexpr int kMD   = 4;                // window radius (WS=9)
constexpr int kW2   = 81;
constexpr int kHid  = 512;              // CPB hidden
}

__device__ __forceinline__ uint32_t bf16_bits(float x) {
  union { __hip_bfloat16 h; uint16_t u; } cv;
  cv.h = __float2bfloat16(x);
  return (uint32_t)cv.u;
}

// ---------------------------------------------------------------------------
// f32 -> bf16 repack (activations)
// ---------------------------------------------------------------------------
__global__ void __launch_bounds__(256)
f32_to_bf16_kernel(const float* __restrict__ in, __hip_bfloat16* __restrict__ out, int n) {
  int i = blockIdx.x * 256 + threadIdx.x;
  if (i < n) out[i] = __float2bfloat16(in[i]);
}

// ---------------------------------------------------------------------------
// Pack f32 weight (K x N, row-major) into WMMA B-fragment order (bf16 pairs).
// Layout: dword[ ((ct*KT + kt)*32 + lane)*8 + i ]
//   ct = 16-col tile, kt = 32-k tile, lane holds col = ct*16 + (lane&15),
//   K = kt*32 + (lane>>4)*8 + 2i + (i>=4 ? 8 : 0) {+0,+1}   (dense 32x16 bf16 B layout)
// GEMM then reads 8 contiguous dwords per lane per K-step (2 x b128).
// ---------------------------------------------------------------------------
__global__ void __launch_bounds__(256)
pack_wmma_b_kernel(const float* __restrict__ W, uint32_t* __restrict__ out, int N, int K) {
  int t = blockIdx.x * 256 + threadIdx.x;
  int total = (N >> 4) * (K >> 5) * 32;
  if (t >= total) return;
  int lane = t & 31;
  int tmp  = t >> 5;
  int KT   = K >> 5;
  int kt   = tmp % KT;
  int ct   = tmp / KT;
  int col   = ct * 16 + (lane & 15);
  int kbase = kt * 32 + (lane >> 4) * 8;
  uint32_t* o = out + (size_t)t * 8;
#pragma unroll
  for (int i = 0; i < 8; ++i) {
    int k0 = kbase + 2 * i + (i >= 4 ? 8 : 0);
    o[i] = bf16_bits(W[(size_t)k0 * N + col]) |
           (bf16_bits(W[(size_t)(k0 + 1) * N + col]) << 16);
  }
}

// ---------------------------------------------------------------------------
// CPB relative-position-bias MLP: (81,2) -> 512 relu -> 4 -> 16*sigmoid
// biasTab layout: [nh][d]  (nh-major, d = (dy+4)*9 + (dx+4))
// ---------------------------------------------------------------------------
__device__ __forceinline__ float cpb_coord(int v) {
  float t = 2.0f * (float)v - 8.0f;                       // in [-8, 8]
  return copysignf(log2f(fabsf(t) + 1.0f) * (1.0f / 3.0f), t);
}

__global__ void __launch_bounds__(352)
cpb_bias_kernel(const float* __restrict__ w1, const float* __restrict__ b1,
                const float* __restrict__ w2, float* __restrict__ biasTab) {
  int t = threadIdx.x;
  if (t >= kW2 * kNH) return;
  int d  = t >> 2;
  int nh = t & 3;
  float c0 = cpb_coord(d / 9);
  float c1 = cpb_coord(d % 9);
  float acc = 0.0f;
  for (int j = 0; j < kHid; ++j) {
    float hv = fmaxf(fmaf(c0, w1[j], fmaf(c1, w1[kHid + j], b1[j])), 0.0f);
    acc = fmaf(hv, w2[j * kNH + nh], acc);
  }
  biasTab[nh * kW2 + d] = 16.0f / (1.0f + __expf(-acc));
}

// ---------------------------------------------------------------------------
// L2-normalize 32-channel groups (per pixel, per head).  ngroups = rows*NH.
// ---------------------------------------------------------------------------
__global__ void __launch_bounds__(256)
l2norm_kernel(float* __restrict__ base, int ngroups, int rowStride) {
  int g = blockIdx.x * 256 + threadIdx.x;
  if (g >= ngroups) return;
  float* p = base + (size_t)(g >> 2) * rowStride + (g & 3) * kHC;
  float s = 0.0f;
#pragma unroll
  for (int c = 0; c < kHC; ++c) s = fmaf(p[c], p[c], s);
  float inv = 1.0f / fmaxf(sqrtf(s), 1e-12f);
#pragma unroll
  for (int c = 0; c < kHC; ++c) p[c] *= inv;
}

// ---------------------------------------------------------------------------
// bf16 WMMA GEMM:  out[M,N] = act( A[M,K] @ W[K,N] + bias )
//   N, K compile-time; M % 128 == 0; N % 64 == 0.
//   8 waves / block, each wave owns a 16x64 tile: 4 accumulators, A fragment
//   loaded once per K-step (2 x b128) and reused for 4 packed-B tiles
//   (2 x b128 each) -> 4 v_wmma per K-step, 0.5 A-load per wmma.
//   ACT: 0 = none, 1 = exact GELU.  OUT_BF16 selects output type.
// ---------------------------------------------------------------------------
template <int N, int K, int ACT, bool OUT_BF16>
__global__ void __launch_bounds__(256)
wmma_gemm_kernel(const __hip_bfloat16* __restrict__ A,
                 const uint32_t* __restrict__ Bp,
                 const float* __restrict__ bias,
                 void* __restrict__ out, int M) {
  constexpr int KT = K / 32;
  const int lane = threadIdx.x & 31;
  const int wave = threadIdx.x >> 5;
  const int tileRow = blockIdx.y * 128 + wave * 16;
  const int ct0     = blockIdx.x * 4;                  // first 16-column tile
  if (tileRow >= M) return;                            // wave-uniform

  const int r    = lane & 15;
  const int half = lane >> 4;

  v8f acc[4] = {};
  // A fragment: per lane two contiguous 16B spans at k0+half*8 and k0+half*8+16
  const __hip_bfloat16* arow = A + (size_t)(tileRow + r) * K + half * 8;
  // B fragments: 8 contiguous dwords per lane per K-tile; consecutive column
  // tiles are KT*256 dwords apart (compile-time constant).
  const uint32_t* brow = Bp + ((size_t)ct0 * KT * 32 + lane) * 8;

  union Frag { uint32_t u[8]; v16bf v; uint4 q[2]; };

#pragma unroll
  for (int kt = 0; kt < KT; ++kt) {
    __builtin_prefetch(arow + kt * 32 + 32, 0, 1);   // next A k-tile (speculative)
    Frag a;
    a.q[0] = *reinterpret_cast<const uint4*>(arow + kt * 32);
    a.q[1] = *reinterpret_cast<const uint4*>(arow + kt * 32 + 16);
#pragma unroll
    for (int j = 0; j < 4; ++j) {
      Frag b;
      const uint32_t* bj = brow + (size_t)j * (KT * 256) + kt * 256;
      b.q[0] = *reinterpret_cast<const uint4*>(bj);
      b.q[1] = *reinterpret_cast<const uint4*>(bj + 4);
      acc[j] = __builtin_amdgcn_wmma_f32_16x16x32_bf16(
          /*neg_a=*/false, a.v, /*neg_b=*/false, b.v,
          /*c_mod=*/(short)0, acc[j], /*reuse_a=*/false, /*reuse_b=*/false);
    }
  }

  // D layout: VGPR i -> row tileRow + half*8 + i, col = (ct0+j)*16 + r.
  // Single base pointer; all store offsets are immediates.
  const int col0 = ct0 * 16 + r;
  if (OUT_BF16) {
    __hip_bfloat16* op =
        reinterpret_cast<__hip_bfloat16*>(out) + (size_t)(tileRow + half * 8) * N + col0;
#pragma unroll
    for (int j = 0; j < 4; ++j) {
      const float bv = bias ? bias[col0 + j * 16] : 0.0f;
#pragma unroll
      for (int i = 0; i < 8; ++i) {
        float v = acc[j][i] + bv;
        if (ACT == 1) v = 0.5f * v * (1.0f + erff(v * 0.70710678118654752f));
        op[i * N + j * 16] = __float2bfloat16(v);
      }
    }
  } else {
    float* op = reinterpret_cast<float*>(out) + (size_t)(tileRow + half * 8) * N + col0;
#pragma unroll
    for (int j = 0; j < 4; ++j) {
      const float bv = bias ? bias[col0 + j * 16] : 0.0f;
#pragma unroll
      for (int i = 0; i < 8; ++i) {
        float v = acc[j][i] + bv;
        if (ACT == 1) v = 0.5f * v * (1.0f + erff(v * 0.70710678118654752f));
        op[i * N + j * 16] = v;
      }
    }
  }
}

// ---------------------------------------------------------------------------
// Bilateral window attention with online softmax.
//   logit[d] = q(p) . (k0(p-d) + k1(p+d));  attn = softmax(logit*scale + bias)
//   xb = sum_d attn[d] * v0(p-d);  xf = sum_d attn[d] * v1(p+d)
//   Writes xcat (b, n, 2C) in bf16 directly for the proj GEMM.
//   One thread per (batch, head, pixel); 96 accumulator VGPRs per lane.
// ---------------------------------------------------------------------------
__global__ void __launch_bounds__(256)
bi_window_attn_kernel(const float* __restrict__ q,
                      const float* __restrict__ kv0,
                      const float* __restrict__ kv1,
                      const float* __restrict__ logit_scale,
                      const float* __restrict__ biasTab,
                      __hip_bfloat16* __restrict__ xcat) {
  int g = blockIdx.x * 256 + threadIdx.x;           // [0, B*NH*Npix)
  int pix  = g % kNpix;
  int hn   = g / kNpix;
  int head = hn & (kNH - 1);
  int bb   = hn >> 2;
  int py = pix / kW, px = pix % kW;

  const size_t rowQ = (size_t)(bb * kNpix + pix);
  const float* qp = q + rowQ * kC + head * kHC;
  float qr[kHC];
#pragma unroll
  for (int c = 0; c < kHC; ++c) qr[c] = qp[c];

  const float scale = __expf(fminf(logit_scale[head], 4.6051701860f)); // clamp log(100)
  const float* bt = biasTab + head * kW2;

  float m = -INFINITY, l = 0.0f;
  float xb[kHC], xf[kHC];
#pragma unroll
  for (int c = 0; c < kHC; ++c) { xb[c] = 0.0f; xf[c] = 0.0f; }

  int di = 0;
  for (int dy = -kMD; dy <= kMD; ++dy) {
    for (int dx = -kMD; dx <= kMD; ++dx, ++di) {
      int ym = py - dy, xm = px - dx;      // p - d  (k0/v0 side)
      int yp = py + dy, xp = px + dx;      // p + d  (k1/v1 side)
      bool ok0 = (ym >= 0) & (ym < kH) & (xm >= 0) & (xm < kW);
      bool ok1 = (yp >= 0) & (yp < kH) & (xp >= 0) & (xp < kW);

      const float* k0p = kv0 + ((size_t)(bb * kNpix + ym * kW + xm)) * (2 * kC) + head * kHC;
      const float* k1p = kv1 + ((size_t)(bb * kNpix + yp * kW + xp)) * (2 * kC) + head * kHC;

      float logit = 0.0f;
#pragma unroll
      for (int c = 0; c < kHC; ++c) {
        float ks = 0.0f;
        if (ok0) ks += k0p[c];
        if (ok1) ks += k1p[c];
        logit = fmaf(qr[c], ks, logit);
      }
      float t = fmaf(logit, scale, bt[di]);

      // online softmax update
      float nm   = fmaxf(m, t);
      float corr = __expf(m - nm);     // exp(-inf) = 0 on first iteration
      float p    = __expf(t - nm);
      m = nm;
      l = fmaf(l, corr, p);

      const float* v0p = k0p + kC;     // v is the second half of the kv row
      const float* v1p = k1p + kC;
#pragma unroll
      for (int c = 0; c < kHC; ++c) {
        float a0 = ok0 ? v0p[c] : 0.0f;
        float a1 = ok1 ? v1p[c] : 0.0f;
        xb[c] = fmaf(xb[c], corr, p * a0);
        xf[c] = fmaf(xf[c], corr, p * a1);
      }
    }
  }

  float inv = 1.0f / l;
  __hip_bfloat16* o = xcat + rowQ * (2 * kC);
#pragma unroll
  for (int c = 0; c < kHC; ++c) {
    o[head * kHC + c]      = __float2bfloat16(xb[c] * inv);
    o[kC + head * kHC + c] = __float2bfloat16(xf[c] * inv);
  }
}

// ---------------------------------------------------------------------------
// Row LayerNorm (C=128, eps=1e-6) + residual:  out = res + LN(z)*w + b
//   One wave per row (wave32: 4 elems/lane, shfl_xor reduction).
//   Optionally also emits a bf16 copy of the result for the next GEMM.
// ---------------------------------------------------------------------------
__global__ void __launch_bounds__(256)
ln_residual_kernel(const float* __restrict__ z, const float* __restrict__ res,
                   const float* __restrict__ w, const float* __restrict__ b,
                   float* __restrict__ outF, __hip_bfloat16* __restrict__ outBf,
                   int nrows) {
  int lane = threadIdx.x & 31;
  int row  = blockIdx.x * 8 + (threadIdx.x >> 5);
  if (row >= nrows) return;
  const float* zr = z + (size_t)row * kC;

  float v[4], s = 0.0f, s2 = 0.0f;
#pragma unroll
  for (int j = 0; j < 4; ++j) {
    v[j] = zr[lane + 32 * j];
    s  += v[j];
    s2  = fmaf(v[j], v[j], s2);
  }
#pragma unroll
  for (int off = 16; off > 0; off >>= 1) {
    s  += __shfl_xor(s,  off, 32);
    s2 += __shfl_xor(s2, off, 32);
  }
  float mu  = s * (1.0f / kC);
  float var = fmaf(-mu, mu, s2 * (1.0f / kC));
  float inv = rsqrtf(var + 1e-6f);

#pragma unroll
  for (int j = 0; j < 4; ++j) {
    int c = lane + 32 * j;
    float y = fmaf((v[j] - mu) * inv, w[c], b[c]) + res[(size_t)row * kC + c];
    outF[(size_t)row * kC + c] = y;
    if (outBf) outBf[(size_t)row * kC + c] = __float2bfloat16(y);
  }
}

// ---------------------------------------------------------------------------
// Launch: pack weights -> q/kv GEMMs -> l2norm -> window attn -> proj ->
//         LN+res -> fc1(GELU) -> fc2 -> LN+res.  Workspace ~283.5 MB, aliased.
// ---------------------------------------------------------------------------
extern "C" void kernel_launch(void* const* d_in, const int* in_sizes, int n_in,
                              void* d_out, int out_size, void* d_ws, size_t ws_size,
                              hipStream_t stream) {
  const float* x0   = (const float*)d_in[0];
  const float* x1   = (const float*)d_in[1];
  const float* xt   = (const float*)d_in[2];
  const float* q_w  = (const float*)d_in[3];
  const float* q_b  = (const float*)d_in[4];
  const float* kv_w = (const float*)d_in[5];
  const float* kv_b = (const float*)d_in[6];
  const float* lsc  = (const float*)d_in[7];
  const float* cw1  = (const float*)d_in[8];
  const float* cb1  = (const float*)d_in[9];
  const float* cw2  = (const float*)d_in[10];
  const float* pr_w = (const float*)d_in[11];
  const float* pr_b = (const float*)d_in[12];
  const float* n1w  = (const float*)d_in[13];
  const float* n1b  = (const float*)d_in[14];
  const float* f1w  = (const float*)d_in[15];
  const float* f1b  = (const float*)d_in[16];
  const float* f2w  = (const float*)d_in[17];
  const float* f2b  = (const float*)d_in[18];
  const float* n2w  = (const float*)d_in[19];
  const float* n2b  = (const float*)d_in[20];
  float* out = (float*)d_out;

  char* ws = (char*)d_ws;
  size_t off = 0;
  auto alloc = [&](size_t bytes) {
    size_t o = off;
    off = (off + bytes + 255) & ~(size_t)255;
    return o;
  };

  const size_t szActBf = (size_t)kBN * kC * sizeof(__hip_bfloat16);       // 18.9 MB
  const size_t szRowF  = (size_t)kBN * kC * sizeof(float);                // 37.7 MB
  const size_t szKV    = (size_t)kBN * 2 * kC * sizeof(float);            // 75.5 MB
  const size_t szXcat  = (size_t)kBN * 2 * kC * sizeof(__hip_bfloat16);   // 37.7 MB

  __hip_bfloat16* xt_bf = (__hip_bfloat16*)(ws + alloc(szActBf));
  __hip_bfloat16* x0_bf = (__hip_bfloat16*)(ws + alloc(szActBf));
  __hip_bfloat16* x1_bf = (__hip_bfloat16*)(ws + alloc(szActBf));
  float* qbuf = (float*)(ws + alloc(szRowF));
  float* kv0  = (float*)(ws + alloc(szKV));
  float* kv1  = (float*)(ws + alloc(szKV));
  __hip_bfloat16* xcat = (__hip_bfloat16*)(ws + alloc(szXcat));
  // Packed-fragment weights (dword arrays, same byte size as bf16 weights)
  uint32_t* wq  = (uint32_t*)(ws + alloc((size_t)kC * kC * 2));
  uint32_t* wkv = (uint32_t*)(ws + alloc((size_t)kC * 2 * kC * 2));
  uint32_t* wpr = (uint32_t*)(ws + alloc((size_t)2 * kC * kC * 2));
  uint32_t* wf1 = (uint32_t*)(ws + alloc((size_t)kC * 4 * kC * 2));
  uint32_t* wf2 = (uint32_t*)(ws + alloc((size_t)4 * kC * kC * 2));
  float* biasTab = (float*)(ws + alloc((size_t)kNH * kW2 * sizeof(float)));

  // Aliases over buffers that are dead by the time these are written:
  float* projOut = qbuf;                                          // q dead after attn
  float* xres    = kv0;                                           // kv0 dead after attn
  __hip_bfloat16* xres_bf = (__hip_bfloat16*)((char*)kv0 + szRowF);
  __hip_bfloat16* hmid_bf = (__hip_bfloat16*)kv1;                 // 75.5 MB fits exactly
  float* zbuf    = (float*)xcat;                                  // xcat dead after proj

  const int nAct = kBN * kC;             // 9,437,184
  const int cvtB = (nAct + 255) / 256;   // 36864 blocks

  // --- small precompute -----------------------------------------------------
  cpb_bias_kernel<<<1, 352, 0, stream>>>(cw1, cb1, cw2, biasTab);
  f32_to_bf16_kernel<<<cvtB, 256, 0, stream>>>(xt, xt_bf, nAct);
  f32_to_bf16_kernel<<<cvtB, 256, 0, stream>>>(x0, x0_bf, nAct);
  f32_to_bf16_kernel<<<cvtB, 256, 0, stream>>>(x1, x1_bf, nAct);
  // pack weights into WMMA fragment order: total threads = (N/16)*(K/32)*32
  pack_wmma_b_kernel<<<4,  256, 0, stream>>>(q_w,  wq,  kC,     kC);      // 1024
  pack_wmma_b_kernel<<<8,  256, 0, stream>>>(kv_w, wkv, 2 * kC, kC);      // 2048
  pack_wmma_b_kernel<<<8,  256, 0, stream>>>(pr_w, wpr, kC,     2 * kC);  // 2048
  pack_wmma_b_kernel<<<16, 256, 0, stream>>>(f1w,  wf1, 4 * kC, kC);      // 4096
  pack_wmma_b_kernel<<<16, 256, 0, stream>>>(f2w,  wf2, kC,     4 * kC);  // 4096

  const dim3 blk(256);
  const int rowTiles = kBN / 128;        // 576 (8 waves x 16 rows per block)

  // --- projections (WMMA) ---------------------------------------------------
  wmma_gemm_kernel<128, 128, 0, false><<<dim3(2, rowTiles), blk, 0, stream>>>(
      xt_bf, wq, q_b, qbuf, kBN);
  wmma_gemm_kernel<256, 128, 0, false><<<dim3(4, rowTiles), blk, 0, stream>>>(
      x0_bf, wkv, kv_b, kv0, kBN);
  wmma_gemm_kernel<256, 128, 0, false><<<dim3(4, rowTiles), blk, 0, stream>>>(
      x1_bf, wkv, kv_b, kv1, kBN);

  // --- cosine-attention normalization --------------------------------------
  const int ngroups = kBN * kNH;         // 294912
  const int l2B = (ngroups + 255) / 256;
  l2norm_kernel<<<l2B, 256, 0, stream>>>(qbuf, ngroups, kC);
  l2norm_kernel<<<l2B, 256, 0, stream>>>(kv0, ngroups, 2 * kC);
  l2norm_kernel<<<l2B, 256, 0, stream>>>(kv1, ngroups, 2 * kC);

  // --- bilateral window attention ------------------------------------------
  const int nAttn = kB * kNH * kNpix;    // 294912
  bi_window_attn_kernel<<<(nAttn + 255) / 256, 256, 0, stream>>>(
      qbuf, kv0, kv1, lsc, biasTab, xcat);

  // --- proj + LN + residual -------------------------------------------------
  wmma_gemm_kernel<128, 256, 0, false><<<dim3(2, rowTiles), blk, 0, stream>>>(
      xcat, wpr, pr_b, projOut, kBN);
  ln_residual_kernel<<<kBN / 8, 256, 0, stream>>>(
      projOut, xt, n1w, n1b, xres, xres_bf, kBN);

  // --- MLP: fc1 (GELU, bf16 out) -> fc2 -> LN + residual --------------------
  wmma_gemm_kernel<512, 128, 1, true><<<dim3(8, rowTiles), blk, 0, stream>>>(
      xres_bf, wf1, f1b, hmid_bf, kBN);
  wmma_gemm_kernel<128, 512, 0, false><<<dim3(2, rowTiles), blk, 0, stream>>>(
      hmid_bf, wf2, f2b, zbuf, kBN);
  ln_residual_kernel<<<kBN / 8, 256, 0, stream>>>(
      zbuf, xres, n2w, n2b, out, nullptr, kBN);

  (void)in_sizes; (void)n_in; (void)out_size; (void)ws_size;
}
